// GNN_88003879895124
// MI455X (gfx1250) — compile-verified, matrix operand order checked by reading, not run on hardware
//
#include <hip/hip_runtime.h>
#include <math.h>

typedef unsigned short u16;
typedef __bf16 bf16_t;
typedef bf16_t v16bf __attribute__((ext_vector_type(16)));
typedef bf16_t v8bf  __attribute__((ext_vector_type(8)));
typedef float  v8f   __attribute__((ext_vector_type(8)));

#define NNODES 12288
#define NGRAPH 384
#define NELEC  32
#define F0     672          // input features
#define F1     400          // layer-1 out
#define K1     1344         // 672 (agg) + 672 (root)
#define K2     800          // 400 + 400
#define N1PAD  448          // 400 padded to 7*64
#define N2PAD  320          // 304 padded to 5*64
#define F2P    304          // layer-2 out padded storage width
#define F2     300

// ---------- small helpers ----------
__device__ __forceinline__ u16 f2bf(float f) {
    unsigned int u = __float_as_uint(f);
    unsigned int r = 0x7FFFu + ((u >> 16) & 1u);     // round-to-nearest-even
    return (u16)((u + r) >> 16);
}
__device__ __forceinline__ float bf2f(u16 h) {
    return __uint_as_float(((unsigned int)h) << 16);
}
// 16x32 bf16 fragment: lane L -> row (L&15), K halves at kg*8 and kg*8+16
__device__ __forceinline__ v16bf load_frag(const u16* p) {
    union { v16bf v; v8bf h[2]; } u;
    u.h[0] = *(const v8bf*)(p);
    u.h[1] = *(const v8bf*)(p + 16);
    return u.v;
}

// ---------- CDNA5 async global->LDS copy (ASYNCcnt-tracked DMA) ----------
// LDS operand = low 32 bits of the flat shared-memory address (ISA 10.2: LDS
// aperture addresses truncate to the LDS byte offset).
__device__ __forceinline__ void async_g2l_b128(void* ldsdst, const void* gsrc) {
    unsigned l = (unsigned)(unsigned long long)ldsdst;
    asm volatile("global_load_async_to_lds_b128 %0, %1, off"
                 :: "v"(l), "v"(gsrc) : "memory");
}
__device__ __forceinline__ void wait_async0() {
    asm volatile("s_wait_asynccnt 0x0" ::: "memory");
}

// ---------- weight pack: wcat[n, 0:Ka]=wa[n], wcat[n, Ka:Ka+Kb]=wb[n]; pad rows zero ----------
__global__ void pack_w_kernel(u16* __restrict__ dst,
                              const float* __restrict__ wa, const float* __restrict__ wb,
                              int Nreal, int Ka, int Kb, int total) {
    int idx = blockIdx.x * blockDim.x + threadIdx.x;
    if (idx >= total) return;
    int Kt = Ka + Kb;
    int n = idx / Kt, k = idx - n * Kt;
    float v = 0.f;
    if (n < Nreal) v = (k < Ka) ? wa[(size_t)n * Ka + k] : wb[(size_t)n * Kb + (k - Ka)];
    dst[idx] = f2bf(v);
}

// ---------- layer-1 aggregation: acat1[:,0:672]=A^T x  (bf16), acat1[:,672:1344]=x (bf16) ----------
__global__ void agg1_kernel(const float* __restrict__ x, const float* __restrict__ ea,
                            u16* __restrict__ acat1) {
    __shared__ float sA[NELEC * NELEC];       // 4 KB
    __shared__ u16   sX[NELEC * F0];          // 42 KB
    const int g = blockIdx.x;
    const int t = threadIdx.x;                 // 256 threads = 8 waves
    const float* Ag = ea + (size_t)g * (NELEC * NELEC);
    // adjacency block: raw f32 copy -> async DMA, 16B per lane
    async_g2l_b128(&sA[t * 4], Ag + t * 4);
    const int nb = g * NELEC;
    const float* xg = x + (size_t)nb * F0;
    u16* arow = acat1 + (size_t)nb * K1;
    for (int i = t; i < NELEC * F0; i += 256) {
        int r = i / F0, c = i - r * F0;
        u16 b = f2bf(xg[i]);                   // f32 -> bf16: must go through VALU
        sX[i] = b;
        arow[(size_t)r * K1 + F0 + c] = b;     // root half = x
    }
    wait_async0();
    __syncthreads();
    const int j  = t & 31;                     // dst node (lane) -> sX reads broadcast per wave
    const int cg = t >> 5;                     // feature phase
    float Aj[NELEC];
    #pragma unroll
    for (int i = 0; i < NELEC; ++i) Aj[i] = sA[i * NELEC + j];   // agg[j]=sum_i A[i,j]x[i]
    for (int c = cg; c < F0; c += 8) {
        float acc = 0.f;
        #pragma unroll
        for (int i = 0; i < NELEC; ++i) acc += Aj[i] * bf2f(sX[i * F0 + c]);
        arow[(size_t)j * K1 + c] = f2bf(acc);
    }
}

// ---------- layer-2 aggregation: reads h1 = acat2[:,400:800], writes acat2[:,0:400] ----------
__global__ void agg2_kernel(const float* __restrict__ ea, u16* __restrict__ acat2) {
    __shared__ float sA[NELEC * NELEC];       // 4 KB
    __shared__ u16   sH[NELEC * F1];          // 25.6 KB
    const int g = blockIdx.x;
    const int t = threadIdx.x;
    const float* Ag = ea + (size_t)g * (NELEC * NELEC);
    async_g2l_b128(&sA[t * 4], Ag + t * 4);
    const int nb = g * NELEC;
    u16* arow = acat2 + (size_t)nb * K2;
    // h1 tile: raw bf16 copy -> async DMA, 50 x 16B chunks per row
    for (int c = t; c < NELEC * 50; c += 256) {
        int r = c / 50, o = c - r * 50;        // row, 16B-chunk within row
        async_g2l_b128(&sH[r * F1 + o * 8], arow + (size_t)r * K2 + F1 + o * 8);
    }
    wait_async0();
    __syncthreads();
    const int j  = t & 31;
    const int cg = t >> 5;
    float Aj[NELEC];
    #pragma unroll
    for (int i = 0; i < NELEC; ++i) Aj[i] = sA[i * NELEC + j];
    for (int c = cg; c < F1; c += 8) {
        float acc = 0.f;
        #pragma unroll
        for (int i = 0; i < NELEC; ++i) acc += Aj[i] * bf2f(sH[i * F1 + c]);
        arow[(size_t)j * K2 + c] = f2bf(acc);
    }
}

// ---------- bf16 WMMA GEMM: C = tanh(A @ B^T + bias) ----------
// A: [12288, K] bf16 row-major; B: [Npad, K] bf16 row-major (weights, zero-padded rows).
// block = 256 threads (8 waves); tile = 128 rows x 64 cols; wave: 16 rows x 4 n-subtiles.
// K is a template constant so all subtile offsets fold into load immediates; explicit
// double-buffering keeps next-step loads in flight under the current WMMA group.
template<int K, bool BF16OUT>
__global__ void gemm_kernel(const u16* __restrict__ A, const u16* __restrict__ B,
                            const float* __restrict__ bias, int Nout, int Nbias,
                            u16* __restrict__ Cb, float* __restrict__ Cf,
                            int ldc, int coff) {
    const int lane = threadIdx.x & 31;
    const int wv   = threadIdx.x >> 5;
    const int row0 = blockIdx.x * 128 + wv * 16;
    const int col0 = blockIdx.y * 64;
    const int frow = lane & 15;
    const int kg   = lane >> 4;

    const u16* ap = A + (size_t)(row0 + frow) * K + kg * 8;
    const u16* bp = B + (size_t)(col0 + frow) * K + kg * 8;

    v8f acc[4] = {{}, {}, {}, {}};
    v16bf af = load_frag(ap);
    v16bf bf[4];
    #pragma unroll
    for (int t = 0; t < 4; ++t) bf[t] = load_frag(bp + t * 16 * K);

    #pragma unroll 2
    for (int k = 32; k < K; k += 32) {
        v16bf afn = load_frag(ap + k);
        v16bf bfn[4];
        #pragma unroll
        for (int t = 0; t < 4; ++t) bfn[t] = load_frag(bp + t * 16 * K + k);
        __builtin_prefetch(ap + k + 256, 0, 1);          // global_prefetch ahead on A
        #pragma unroll
        for (int t = 0; t < 4; ++t)
            acc[t] = __builtin_amdgcn_wmma_f32_16x16x32_bf16(
                         false, af, false, bf[t], (short)0, acc[t], false, false);
        af = afn;
        #pragma unroll
        for (int t = 0; t < 4; ++t) bf[t] = bfn[t];
    }
    #pragma unroll
    for (int t = 0; t < 4; ++t)
        acc[t] = __builtin_amdgcn_wmma_f32_16x16x32_bf16(
                     false, af, false, bf[t], (short)0, acc[t], false, false);

    // C/D layout: lane n(0..15) col, vgpr v -> row v + 8*(lane>>4)
    const int n16 = lane & 15;
    const int mh  = lane >> 4;
    #pragma unroll
    for (int t = 0; t < 4; ++t) {
        int n = col0 + t * 16 + n16;
        if (n >= Nout) continue;
        float bv = (n < Nbias) ? bias[n] : 0.f;
        #pragma unroll
        for (int v = 0; v < 8; ++v) {
            int m = row0 + v + 8 * mh;
            float r = tanhf(acc[t][v] + bv);
            if (BF16OUT) Cb[(size_t)m * ldc + coff + n] = f2bf(r);
            else         Cf[(size_t)m * ldc + n] = r;
        }
    }
}

// ---------- global add pool: g[graph,c] = sum over 32 nodes of h2 ----------
__global__ void pool_kernel(const float* __restrict__ h2, float* __restrict__ gout) {
    const int g = blockIdx.x;
    const int c = threadIdx.x;
    if (c >= F2) return;
    const float* p = h2 + (size_t)g * NELEC * F2P + c;
    float s = 0.f;
    #pragma unroll
    for (int i = 0; i < NELEC; ++i) s += p[(size_t)i * F2P];
    gout[(size_t)g * F2 + c] = s;
}

// ---------- fused conv1d x3 + MLP head, one block per batch sample ----------
__global__ void head_kernel(const float* __restrict__ gp,
                            const float* __restrict__ cw1, const float* __restrict__ cb1,
                            const float* __restrict__ cw2, const float* __restrict__ cb2,
                            const float* __restrict__ cw3, const float* __restrict__ cb3,
                            const float* __restrict__ mw1, const float* __restrict__ mb1,
                            const float* __restrict__ mw2, const float* __restrict__ mb2,
                            const float* __restrict__ mw3, const float* __restrict__ mb3,
                            float* __restrict__ out) {
    __shared__ float sin[12 * 300];
    __shared__ float s1[8 * 148];
    __shared__ float s2[4 * 72];
    __shared__ float s3[68];
    __shared__ float s4[32];
    __shared__ float s5[16];
    const int b = blockIdx.x;
    const int t = threadIdx.x;                 // 256
    const float* gb = gp + (size_t)b * 12 * 300;
    for (int i = t; i < 12 * 300; i += 256) sin[i] = gb[i];
    __syncthreads();
    // conv1: 12ch,len300 -> 8ch,len148, k=5, stride 2, relu
    for (int i = t; i < 8 * 148; i += 256) {
        int oc = i / 148, o = i - oc * 148;
        float a = cb1[oc];
        #pragma unroll
        for (int ic = 0; ic < 12; ++ic)
            #pragma unroll
            for (int k = 0; k < 5; ++k)
                a += sin[ic * 300 + 2 * o + k] * cw1[(oc * 12 + ic) * 5 + k];
        s1[i] = fmaxf(a, 0.f);
    }
    __syncthreads();
    // conv2: 8->4, len148->72, k=5, stride 2, relu
    for (int i = t; i < 4 * 72; i += 256) {
        int oc = i / 72, o = i - oc * 72;
        float a = cb2[oc];
        #pragma unroll
        for (int ic = 0; ic < 8; ++ic)
            #pragma unroll
            for (int k = 0; k < 5; ++k)
                a += s1[ic * 148 + 2 * o + k] * cw2[(oc * 8 + ic) * 5 + k];
        s2[i] = fmaxf(a, 0.f);
    }
    __syncthreads();
    // conv3: 4->1, len72->68, k=5, stride 1, relu
    for (int o = t; o < 68; o += 256) {
        float a = cb3[0];
        #pragma unroll
        for (int ic = 0; ic < 4; ++ic)
            #pragma unroll
            for (int k = 0; k < 5; ++k)
                a += s2[ic * 72 + o + k] * cw3[ic * 5 + k];
        s3[o] = fmaxf(a, 0.f);
    }
    __syncthreads();
    // mlp1: 68 -> 32, relu
    for (int o = t; o < 32; o += 256) {
        float a = mb1[o];
        #pragma unroll
        for (int j = 0; j < 68; ++j) a += s3[j] * mw1[o * 68 + j];
        s4[o] = fmaxf(a, 0.f);
    }
    __syncthreads();
    // mlp2: 32 -> 16, relu
    for (int o = t; o < 16; o += 256) {
        float a = mb2[o];
        #pragma unroll
        for (int j = 0; j < 32; ++j) a += s4[j] * mw2[o * 32 + j];
        s5[o] = fmaxf(a, 0.f);
    }
    __syncthreads();
    if (t == 0) {
        float a = mb3[0];
        #pragma unroll
        for (int j = 0; j < 16; ++j) a += s5[j] * mw3[j];
        out[b] = a;
    }
}

// ---------- host launch ----------
extern "C" void kernel_launch(void* const* d_in, const int* in_sizes, int n_in,
                              void* d_out, int out_size, void* d_ws, size_t ws_size,
                              hipStream_t stream) {
    (void)in_sizes; (void)n_in; (void)out_size; (void)ws_size;
    const float* x       = (const float*)d_in[0];
    // d_in[1] = edge_index (structure is known: dense 32-node blocks) -> unused
    const float* ea      = (const float*)d_in[2];
    const float* w_rel1  = (const float*)d_in[3];
    const float* b_rel1  = (const float*)d_in[4];
    const float* w_root1 = (const float*)d_in[5];
    const float* w_rel2  = (const float*)d_in[6];
    const float* b_rel2  = (const float*)d_in[7];
    const float* w_root2 = (const float*)d_in[8];
    const float* cw1 = (const float*)d_in[9],  *cb1 = (const float*)d_in[10];
    const float* cw2 = (const float*)d_in[11], *cb2 = (const float*)d_in[12];
    const float* cw3 = (const float*)d_in[13], *cb3 = (const float*)d_in[14];
    const float* mw1 = (const float*)d_in[15], *mb1 = (const float*)d_in[16];
    const float* mw2 = (const float*)d_in[17], *mb2 = (const float*)d_in[18];
    const float* mw3 = (const float*)d_in[19], *mb3 = (const float*)d_in[20];

    char* ws = (char*)d_ws;
    size_t off = 0;
    auto carve = [&](size_t bytes) -> void* {
        void* p = ws + off;
        off = (off + bytes + 255) & ~(size_t)255;
        return p;
    };
    u16*   acat1 = (u16*)  carve((size_t)NNODES * K1 * 2);      // [agg1 | x] bf16
    u16*   acat2 = (u16*)  carve((size_t)NNODES * K2 * 2);      // [agg2 | h1] bf16
    u16*   wcat1 = (u16*)  carve((size_t)N1PAD * K1 * 2);       // [w_rel1 | w_root1]
    u16*   wcat2 = (u16*)  carve((size_t)N2PAD * K2 * 2);       // [w_rel2 | w_root2]
    float* h2    = (float*)carve((size_t)NNODES * F2P * 4);
    float* gp    = (float*)carve((size_t)NGRAPH * F2 * 4);

    // pack weights (bf16, K-concat, zero-padded rows)
    {
        int tot1 = N1PAD * K1;
        pack_w_kernel<<<(tot1 + 255) / 256, 256, 0, stream>>>(wcat1, w_rel1, w_root1, F1, F0, F0, tot1);
        int tot2 = N2PAD * K2;
        pack_w_kernel<<<(tot2 + 255) / 256, 256, 0, stream>>>(wcat2, w_rel2, w_root2, F2, F1, F1, tot2);
    }
    // layer 1
    agg1_kernel<<<NGRAPH, 256, 0, stream>>>(x, ea, acat1);
    gemm_kernel<K1, true><<<dim3(NNODES / 128, N1PAD / 64), 256, 0, stream>>>(
        acat1, wcat1, b_rel1, F1, F1,
        acat2, nullptr, K2, F1);                  // h1 -> acat2[:, 400:800] (bf16)
    // layer 2
    agg2_kernel<<<NGRAPH, 256, 0, stream>>>(ea, acat2);
    gemm_kernel<K2, false><<<dim3(NNODES / 128, N2PAD / 64), 256, 0, stream>>>(
        acat2, wcat2, b_rel2, F2P, F2,
        nullptr, h2, F2P, 0);                     // h2 f32 [12288, 304]
    // pool + head
    pool_kernel<<<NGRAPH, 320, 0, stream>>>(h2, gp);
    head_kernel<<<32, 256, 0, stream>>>(gp, cw1, cb1, cw2, cb2, cw3, cb3,
                                        mw1, mb1, mw2, mb2, mw3, mb3, (float*)d_out);
}
